// GPTJAttention_30794915512449
// MI455X (gfx1250) — matrix operand order repaired and checked by reading, not compiled, not run on hardware
//
#include <hip/hip_runtime.h>
#include <hip/hip_bf16.h>

// ---------------------------------------------------------------------------
// GPT-J attention for MI455X (gfx1250), wave32 + WMMA f16 (f32 accum).
// Pipeline: f32->f16 convert | 3x GEMM (QKV) | RoPE | flash-attn | GEMM (out)
// GLOBAL_LOAD_ASYNC_TO_LDS_B128 staging (ASYNCcnt) with double-buffered GEMM
// K-tiles; guarded by __has_builtin with sync fallback.
// ---------------------------------------------------------------------------

typedef __attribute__((ext_vector_type(16))) _Float16 v16h;
typedef __attribute__((ext_vector_type(8)))  _Float16 v8h;
typedef __attribute__((ext_vector_type(8)))  float    v8f;
typedef __attribute__((ext_vector_type(4)))  int      v4i;

#define SEQ      4096
#define EMBED    4096
#define HEADS    16
#define HEAD_DIM 256
#define ROT      64
#define NELEM    ((size_t)SEQ * (size_t)EMBED)   // 16M elements per matrix

#if defined(__HIP_DEVICE_COMPILE__) && __has_builtin(__builtin_amdgcn_global_load_async_to_lds_b128)
#define HAVE_ASYNC_LDS 1
#else
#define HAVE_ASYNC_LDS 0
#endif

#if HAVE_ASYNC_LDS
typedef __attribute__((address_space(1))) v4i as1_v4i;   // global int4
typedef __attribute__((address_space(3))) v4i as3_v4i;   // LDS int4

// 16B global -> LDS async copy (tracked by ASYNCcnt, no VGPR data movement)
__device__ __forceinline__ void async_copy_b128(const _Float16* g, _Float16* l) {
  __builtin_amdgcn_global_load_async_to_lds_b128(
      (as1_v4i*)(uintptr_t)g,
      (as3_v4i*)(uint32_t)(uintptr_t)l,   // LDS offset lives in low 32 bits
      0, 0);
}
__device__ __forceinline__ void wait_async0() {
#if __has_builtin(__builtin_amdgcn_s_wait_asynccnt)
  __builtin_amdgcn_s_wait_asynccnt(0);
#else
  asm volatile("s_wait_asynccnt 0x0" ::: "memory");
#endif
}
#else
__device__ __forceinline__ void async_copy_b128(const _Float16* g, _Float16* l) {
  *(v8h*)l = *(const v8h*)g;               // sync fallback (ds_store path)
}
__device__ __forceinline__ void wait_async0() {}
#endif

__device__ __forceinline__ v8f wmma_f16(v16h a, v16h b, v8f c) {
  // D = A(16x32 f16) * B(32x16 f16) + C(16x16 f32)
  return __builtin_amdgcn_wmma_f32_16x16x32_f16(
      false, a, false, b, (short)0, c, false, false);
}

__device__ __forceinline__ v16h pack2(const _Float16* lo, const _Float16* hi) {
  union { v16h v; v8h h[2]; } u;
  u.h[0] = *(const v8h*)lo;
  u.h[1] = *(const v8h*)hi;
  return u.v;
}

// ---------------------------------------------------------------------------
// fp32 -> fp16 conversion (grid-stride)
// ---------------------------------------------------------------------------
__global__ __launch_bounds__(256) void cvt_f32_f16(const float* __restrict__ in,
                                                   _Float16* __restrict__ out,
                                                   size_t n) {
  size_t i = (size_t)blockIdx.x * blockDim.x + threadIdx.x;
  size_t stride = (size_t)gridDim.x * blockDim.x;
  for (; i < n; i += stride) out[i] = (_Float16)in[i];
}

// ---------------------------------------------------------------------------
// C[M,N] = A[M,K] @ W[N,K]^T   (A,W f16 row-major; C f16 or f32)
// Block tile 128x128, K-tile 64, double-buffered async LDS staging.
// 8 waves: 4 (M) x 2 (N), wave tile 32x64 = 8 WMMA accumulators.
// ---------------------------------------------------------------------------
#define KT        64
#define LDT       72                       // padded LDS row stride, 144B = 9*16B
#define GEMM_BUF  (128 * LDT)              // one tile buffer (A or B), elems
#define GEMM_LDS_ELEMS (4 * GEMM_BUF)      // {A,B} x 2 buffers

template <typename OutT>
__global__ __launch_bounds__(256) void gemm_xwt(const _Float16* __restrict__ A,
                                                const _Float16* __restrict__ W,
                                                OutT* __restrict__ C,
                                                int M, int N, int K) {
  extern __shared__ _Float16 gsm[];

  const int tid   = threadIdx.x;
  const int lane  = tid & 31;
  const int wave  = tid >> 5;          // 0..7
  const int waveM = wave & 3;          // 4 waves in M
  const int waveN = wave >> 2;         // 2 waves in N
  const int bM    = blockIdx.y * 128;
  const int bN    = blockIdx.x * 128;
  const int half  = lane >> 4;
  const int l16   = lane & 15;
  const int kb8   = half * 8;          // A-frag K sub-base
  const int kst   = half * 16;         // unused (B-frag uses kb8)

  v8f acc[2][4];
#pragma unroll
  for (int i = 0; i < 2; ++i)
#pragma unroll
    for (int j = 0; j < 4; ++j) acc[i][j] = (v8f){};

  // stage one 128x64 A tile + 128x64 B tile into buffer `buf`
  auto stage = [&](int buf, int k0) {
    _Float16* As = &gsm[buf * 2 * GEMM_BUF];
    _Float16* Bs = As + GEMM_BUF;
#pragma unroll
    for (int i = 0; i < 4; ++i) {
      int c  = tid + i * 256;
      int r  = c >> 3;                 // 8 v8h chunks per row
      int dc = (c & 7) * 8;
      async_copy_b128(&A[(size_t)(bM + r) * K + k0 + dc], &As[r * LDT + dc]);
      async_copy_b128(&W[(size_t)(bN + r) * K + k0 + dc], &Bs[r * LDT + dc]);
    }
  };

  int buf = 0;
  stage(buf, 0);
  wait_async0();
  __syncthreads();

  for (int k0 = 0; k0 < K; k0 += KT) {
    if (k0 + KT < K) stage(buf ^ 1, k0 + KT);   // prefetch next tile (async)

    const _Float16* As = &gsm[buf * 2 * GEMM_BUF];
    const _Float16* Bs = As + GEMM_BUF;
#pragma unroll
    for (int kk = 0; kk < KT; kk += 32) {
      v16h af[2], bf[4];
#pragma unroll
      for (int mt = 0; mt < 2; ++mt) {
        const _Float16* p = &As[(waveM * 32 + mt * 16 + l16) * LDT + kk + kb8];
        af[mt] = pack2(p, p + 16);
      }
#pragma unroll
      for (int nt = 0; nt < 4; ++nt) {
        const _Float16* p = &Bs[(waveN * 64 + nt * 16 + l16) * LDT + kk + kb8];
        bf[nt] = pack2(p, p + 16);
      }
#pragma unroll
      for (int mt = 0; mt < 2; ++mt)
#pragma unroll
        for (int nt = 0; nt < 4; ++nt)
          acc[mt][nt] = wmma_f16(af[mt], bf[nt], acc[mt][nt]);
    }

    wait_async0();        // next tile's async copies done (this wave)
    __syncthreads();      // all waves' copies visible
    buf ^= 1;
  }

  // C layout: VGPR r -> row r + 8*half, col = l16
#pragma unroll
  for (int mt = 0; mt < 2; ++mt)
#pragma unroll
    for (int nt = 0; nt < 4; ++nt)
#pragma unroll
      for (int r = 0; r < 8; ++r) {
        int row = bM + waveM * 32 + mt * 16 + r + 8 * half;
        int col = bN + waveN * 64 + nt * 16 + l16;
        C[(size_t)row * N + col] = (OutT)acc[mt][nt][r];
      }
}

// ---------------------------------------------------------------------------
// RoPE (GPT-J rotate-every-two) applied in place on f16 Q,K; first ROT dims.
// ---------------------------------------------------------------------------
__global__ __launch_bounds__(256) void rope_kernel(_Float16* __restrict__ Q,
                                                   _Float16* __restrict__ Kp) {
  int idx = blockIdx.x * blockDim.x + threadIdx.x;   // S*H*(ROT/2) threads
  if (idx >= SEQ * HEADS * (ROT / 2)) return;
  int i = idx & 31;
  int h = (idx >> 5) & 15;
  int s = idx >> 9;
  float freq = __powf(10000.0f, -(float)(2 * i) / (float)ROT);
  float ang = (float)s * freq;
  float sn = __sinf(ang), cs = __cosf(ang);
  size_t base = (size_t)s * EMBED + h * HEAD_DIM + 2 * i;

  float q0 = (float)Q[base], q1 = (float)Q[base + 1];
  Q[base]     = (_Float16)(q0 * cs - q1 * sn);
  Q[base + 1] = (_Float16)(q1 * cs + q0 * sn);

  float k0 = (float)Kp[base], k1 = (float)Kp[base + 1];
  Kp[base]     = (_Float16)(k0 * cs - k1 * sn);
  Kp[base + 1] = (_Float16)(k1 * cs + k0 * sn);
}

// ---------------------------------------------------------------------------
// Causal flash attention. Grid (32 q-blocks, 16 heads), 256 threads (8 waves).
// Wave owns 16 query rows; keys processed 32 at a time.
//   Q in LDS: 128 x 256  (stride 264)  -- async staged
//   K chunk : 32  x 256  (stride 264)  -- async staged
//   V chunk : transposed 256 x 32 (stride 40) -- sync (transpose on store)
//   P tiles : per-wave 16 x 32 (stride 40)
// ---------------------------------------------------------------------------
#define QS_STRIDE 264
#define VT_STRIDE 40
#define ATTN_LDS_ELEMS (128 * QS_STRIDE + 32 * QS_STRIDE + 256 * VT_STRIDE + 8 * 16 * VT_STRIDE)

__global__ __launch_bounds__(256) void flash_attn(const _Float16* __restrict__ Q,
                                                  const _Float16* __restrict__ Kg,
                                                  const _Float16* __restrict__ Vg,
                                                  _Float16* __restrict__ O) {
  extern __shared__ _Float16 smem[];
  _Float16* Qs = smem;                                   // 128*264
  _Float16* Ks = Qs + 128 * QS_STRIDE;                   // 32*264
  _Float16* Vt = Ks + 32 * QS_STRIDE;                    // 256*40 (d-major)
  _Float16* Ps = Vt + 256 * VT_STRIDE;                   // 8 * 16*40

  const int qb   = blockIdx.x;     // 0..31
  const int h    = blockIdx.y;     // 0..15
  const int tid  = threadIdx.x;
  const int lane = tid & 31;
  const int w    = tid >> 5;       // wave 0..7 -> rows w*16..w*16+15
  const int half = lane >> 4;
  const int l16  = lane & 15;
  const int kb8  = half * 8;
  const int kst  = half * 16;

  // ---- stage Q block (async): 128 rows x 256 d = 4096 v8h chunks, 16/thread
#pragma unroll
  for (int i = 0; i < 16; ++i) {
    int c = tid + i * 256;
    int r = c >> 5;                  // 32 chunks per row
    int dc = (c & 31) * 8;
    async_copy_b128(&Q[(size_t)(qb * 128 + r) * EMBED + h * HEAD_DIM + dc],
                    &Qs[r * QS_STRIDE + dc]);
  }
  wait_async0();
  __syncthreads();

  v8f ot[16];
#pragma unroll
  for (int d = 0; d < 16; ++d) ot[d] = (v8f){};
  float mrow[8], lrow[8];
#pragma unroll
  for (int r = 0; r < 8; ++r) { mrow[r] = -3.0e38f; lrow[r] = 0.0f; }

  const float scale = 0.0625f;       // 1/sqrt(256)
  const int kend = (qb + 1) * 128;   // causal bound, multiple of 32

  for (int kb = 0; kb < kend; kb += 32) {
    // ---- stage K chunk (async) and transposed V chunk (sync): 1024 chunks
#pragma unroll
    for (int i = 0; i < 4; ++i) {
      int c = tid + i * 256;
      int r = c >> 5;                // key within chunk
      int dc = (c & 31) * 8;
      async_copy_b128(&Kg[(size_t)(kb + r) * EMBED + h * HEAD_DIM + dc],
                      &Ks[r * QS_STRIDE + dc]);
      v8h vv = *(const v8h*)&Vg[(size_t)(kb + r) * EMBED + h * HEAD_DIM + dc];
#pragma unroll
      for (int j = 0; j < 8; ++j) Vt[(dc + j) * VT_STRIDE + r] = vv[j];
    }
    wait_async0();
    __syncthreads();

    // ---- S = Q Kt : two 16x16 tiles (keys kb..+15 and kb+16..+31)
    v8f s0 = (v8f){}, s1 = (v8f){};
    const _Float16* qrow = &Qs[(w * 16 + l16) * QS_STRIDE];
#pragma unroll
    for (int dt = 0; dt < 8; ++dt) {
      v16h aq = pack2(qrow + dt * 32 + kb8, qrow + dt * 32 + kb8 + 16);
      const _Float16* k0p = &Ks[l16 * QS_STRIDE + dt * 32 + kb8];
      const _Float16* k1p = &Ks[(16 + l16) * QS_STRIDE + dt * 32 + kb8];
      s0 = wmma_f16(aq, pack2(k0p, k0p + 16), s0);
      s1 = wmma_f16(aq, pack2(k1p, k1p + 16), s1);
    }

    // ---- mask + scale + online softmax (rows live across 16-lane halves)
    float corr[8];
    _Float16* Pw = &Ps[w * 16 * VT_STRIDE];
#pragma unroll
    for (int r = 0; r < 8; ++r) {
      int qrow_g = qb * 128 + w * 16 + r + 8 * half;
      float x0 = s0[r] * scale;
      float x1 = s1[r] * scale;
      if (kb + l16 > qrow_g)      x0 = -1.0e9f;
      if (kb + 16 + l16 > qrow_g) x1 = -1.0e9f;
      float rm = fmaxf(x0, x1);
#pragma unroll
      for (int off = 1; off < 16; off <<= 1) rm = fmaxf(rm, __shfl_xor(rm, off));
      float mnew = fmaxf(mrow[r], rm);
      corr[r] = __expf(mrow[r] - mnew);
      float p0 = __expf(x0 - mnew);
      float p1 = __expf(x1 - mnew);
      float rs = p0 + p1;
#pragma unroll
      for (int off = 1; off < 16; off <<= 1) rs += __shfl_xor(rs, off);
      lrow[r] = lrow[r] * corr[r] + rs;
      mrow[r] = mnew;
      // write P tile (C layout -> LDS, re-read below in A-fragment layout)
      Pw[(r + 8 * half) * VT_STRIDE + l16]      = (_Float16)p0;
      Pw[(r + 8 * half) * VT_STRIDE + 16 + l16] = (_Float16)p1;
    }
#pragma unroll
    for (int d = 0; d < 16; ++d)
#pragma unroll
      for (int r = 0; r < 8; ++r) ot[d][r] *= corr[r];

    // ---- O += P(16x32) @ V(32x256): one WMMA per 16-wide d tile
    const _Float16* prow = &Pw[l16 * VT_STRIDE];
    v16h pA = pack2(prow + kb8, prow + kb8 + 16);
#pragma unroll
    for (int dt = 0; dt < 16; ++dt) {
      const _Float16* vrow = &Vt[(dt * 16 + l16) * VT_STRIDE + kb8];
      ot[dt] = wmma_f16(pA, pack2(vrow, vrow + 16), ot[dt]);
    }
    __syncthreads();
  }

  // ---- epilogue: normalize and store f16 at [s][h*256 + d]
#pragma unroll
  for (int dt = 0; dt < 16; ++dt)
#pragma unroll
    for (int r = 0; r < 8; ++r) {
      int row = qb * 128 + w * 16 + r + 8 * half;
      int col = h * HEAD_DIM + dt * 16 + l16;
      O[(size_t)row * EMBED + col] = (_Float16)(ot[dt][r] / lrow[r]);
    }
}

// ---------------------------------------------------------------------------
// Host-side orchestration
// ---------------------------------------------------------------------------
extern "C" void kernel_launch(void* const* d_in, const int* in_sizes, int n_in,
                              void* d_out, int out_size, void* d_ws, size_t ws_size,
                              hipStream_t stream) {
  const float* hs = (const float*)d_in[0];
  const float* wq = (const float*)d_in[1];
  const float* wk = (const float*)d_in[2];
  const float* wv = (const float*)d_in[3];
  const float* wo = (const float*)d_in[4];
  float* out = (float*)d_out;

  // workspace layout: 8 x 16M f16 regions (256 MB total)
  _Float16* hX = (_Float16*)d_ws;        // hidden f16; later reused for attn out
  _Float16* Wq = hX + NELEM;
  _Float16* Wk = Wq + NELEM;
  _Float16* Wv = Wk + NELEM;
  _Float16* Wo = Wv + NELEM;
  _Float16* Qf = Wo + NELEM;
  _Float16* Kf = Qf + NELEM;
  _Float16* Vf = Kf + NELEM;
  _Float16* X2 = hX;                     // attention output reuses hidden slot

  // 1) convert inputs to f16
  cvt_f32_f16<<<4096, 256, 0, stream>>>(hs, hX, NELEM);
  cvt_f32_f16<<<4096, 256, 0, stream>>>(wq, Wq, NELEM);
  cvt_f32_f16<<<4096, 256, 0, stream>>>(wk, Wk, NELEM);
  cvt_f32_f16<<<4096, 256, 0, stream>>>(wv, Wv, NELEM);
  cvt_f32_f16<<<4096, 256, 0, stream>>>(wo, Wo, NELEM);

  // 2) QKV projections (f16 out; [s][h*256+d] layout == reshape in reference)
  dim3 ggrid(EMBED / 128, SEQ / 128);
  size_t gemm_lds = (size_t)GEMM_LDS_ELEMS * sizeof(_Float16);   // 72 KB
  gemm_xwt<_Float16><<<ggrid, 256, gemm_lds, stream>>>(hX, Wq, Qf, SEQ, EMBED, EMBED);
  gemm_xwt<_Float16><<<ggrid, 256, gemm_lds, stream>>>(hX, Wk, Kf, SEQ, EMBED, EMBED);
  gemm_xwt<_Float16><<<ggrid, 256, gemm_lds, stream>>>(hX, Wv, Vf, SEQ, EMBED, EMBED);

  // 3) RoPE on Q,K (first 64 dims of each head)
  rope_kernel<<<(SEQ * HEADS * (ROT / 2)) / 256, 256, 0, stream>>>(Qf, Kf);

  // 4) causal flash attention
  dim3 agrid(SEQ / 128, HEADS);
  size_t attn_lds = (size_t)ATTN_LDS_ELEMS * sizeof(_Float16);   // ~113 KB
  flash_attn<<<agrid, 256, attn_lds, stream>>>(Qf, Kf, Vf, X2);

  // 5) output projection (f32 out)
  gemm_xwt<float><<<ggrid, 256, gemm_lds, stream>>>(X2, Wo, out, SEQ, EMBED, EMBED);
}
